// LoRALinear_68702296867424
// MI455X (gfx1250) — matrix operand order repaired and checked by reading, not compiled, not run on hardware
//
#include <hip/hip_runtime.h>

#define IN_F  4096
#define OUT_F 4096
#define RANK  16
#define MTOT  8192   // 4 * 2048 rows

typedef __attribute__((ext_vector_type(16))) __bf16 v16bf;
typedef __attribute__((ext_vector_type(8)))  float  v8f;

__device__ __forceinline__ unsigned short f2bf(float f) {
  unsigned int u = __float_as_uint(f);
  u += 0x7FFFu + ((u >> 16) & 1u);   // round-to-nearest-even
  return (unsigned short)(u >> 16);
}

// ---------------- Kernel 1: x (f32) -> bf16, 8 elements / thread ----------------
__global__ __launch_bounds__(256) void cvt_x(const float* __restrict__ x,
                                             unsigned short* __restrict__ xb, int n8) {
  int idx = blockIdx.x * blockDim.x + threadIdx.x;
  if (idx >= n8) return;
  const float4* p = reinterpret_cast<const float4*>(x) + (size_t)idx * 2;
  float4 f0 = p[0], f1 = p[1];
  uint4 o;
  o.x = (unsigned)f2bf(f0.x) | ((unsigned)f2bf(f0.y) << 16);
  o.y = (unsigned)f2bf(f0.z) | ((unsigned)f2bf(f0.w) << 16);
  o.z = (unsigned)f2bf(f1.x) | ((unsigned)f2bf(f1.y) << 16);
  o.w = (unsigned)f2bf(f1.z) | ((unsigned)f2bf(f1.w) << 16);
  reinterpret_cast<uint4*>(xb)[idx] = o;
}

// -------- Kernel 2: W_eff = bf16(W + B@A)  (folds rank-16 LoRA into weights) ----
__global__ __launch_bounds__(256) void fuse_w(const float* __restrict__ W,
                                              const float* __restrict__ A,
                                              const float* __restrict__ B,
                                              unsigned short* __restrict__ wb) {
  int idx = blockIdx.x * blockDim.x + threadIdx.x;
  int i = idx & (IN_F - 1);
  int o = idx >> 12;                                  // IN_F == 4096 == 2^12
  float acc = W[(size_t)o * IN_F + i];
#pragma unroll
  for (int r = 0; r < RANK; ++r)
    acc += B[o * RANK + r] * A[r * IN_F + i];
  wb[idx] = f2bf(acc);
}

// ---------------- Kernel 3: bf16 WMMA GEMM  out = xb @ wbT + bias ----------------
// CTA tile 128(M) x 256(N); 8 wave32 waves in 2(M) x 4(N); each wave 64x64 = 4x4 WMMA
// tiles -> 16 v_wmma per K-step of 32. Global->LDS staging uses CDNA5 async copies
// (global_load_async_to_lds_b128, ASYNCcnt) with double-buffered LDS.
#define LSTR 40   // padded LDS row stride in bf16 elems (80B = 20 banks, conflict-free)

__device__ __forceinline__ void async_ld16(unsigned ldsAddr, unsigned long long gAddr) {
  asm volatile("global_load_async_to_lds_b128 %0, %1, off"
               :: "v"(ldsAddr), "v"(gAddr) : "memory");
}

__global__ __launch_bounds__(256) void lora_gemm(const unsigned short* __restrict__ xb,
                                                 const unsigned short* __restrict__ wb,
                                                 const float* __restrict__ bias,
                                                 float* __restrict__ out) {
  __shared__ unsigned short Xs[2][128 * LSTR];
  __shared__ unsigned short Ws[2][256 * LSTR];

  const int tid   = threadIdx.x;
  const int lane  = tid & 31;
  const int wave  = tid >> 5;
  const int waveM = wave & 1;    // 0..1 -> 64 rows each
  const int waveN = wave >> 1;   // 0..3 -> 64 cols each
  const int mCTA  = blockIdx.y * 128;
  const int nCTA  = blockIdx.x * 256;

  // staging map: X -> (row=tid>>1, 16 elems at (tid&1)*16); W -> (row=tid, 32 elems)
  const unsigned long long xg =
      (unsigned long long)(xb + (size_t)(mCTA + (tid >> 1)) * IN_F + (tid & 1) * 16);
  const unsigned long long wg =
      (unsigned long long)(wb + (size_t)(nCTA + tid) * IN_F);
  const unsigned xl0 = (unsigned)(uintptr_t)&Xs[0][(tid >> 1) * LSTR + (tid & 1) * 16];
  const unsigned xl1 = (unsigned)(uintptr_t)&Xs[1][(tid >> 1) * LSTR + (tid & 1) * 16];
  const unsigned wl0 = (unsigned)(uintptr_t)&Ws[0][tid * LSTR];
  const unsigned wl1 = (unsigned)(uintptr_t)&Ws[1][tid * LSTR];

  // issue 6 async b128 copies for one K-tile (X: 32B/thread, W: 64B/thread)
  auto issueTile = [&](int kt, int buf) {
    const unsigned long long gk = (unsigned long long)(kt * 32) * 2u;  // bytes
    const unsigned xl = buf ? xl1 : xl0;
    const unsigned wl = buf ? wl1 : wl0;
    async_ld16(xl,      xg + gk);
    async_ld16(xl + 16, xg + gk + 16);
    async_ld16(wl,      wg + gk);
    async_ld16(wl + 16, wg + gk + 16);
    async_ld16(wl + 32, wg + gk + 32);
    async_ld16(wl + 48, wg + gk + 48);
  };

  v8f acc[4][4];
  const v8f vz = {0.f, 0.f, 0.f, 0.f, 0.f, 0.f, 0.f, 0.f};
#pragma unroll
  for (int i = 0; i < 4; ++i)
#pragma unroll
    for (int j = 0; j < 4; ++j) acc[i][j] = vz;

  // fragment addressing (ISA 7.12.2, wave32):
  // A 16x32 bf16: lane L -> row (L&15), K-quads at (L>>4)*8 and +16
  // B 32x16 bf16: lane L -> W-row (L&15), K-quads at (L>>4)*8 and +16
  const int frow = lane & 15;
  const int akb  = (lane >> 4) * 8;
  const int bkb  = (lane >> 4) * 8;
  const int NT   = IN_F / 32;

  issueTile(0, 0);

  int buf = 0;
#pragma unroll 1
  for (int kt = 0; kt < NT; ++kt) {
    if (kt + 1 < NT) {
      issueTile(kt + 1, buf ^ 1);
      asm volatile("s_wait_asynccnt 0x6" ::: "memory");  // this tile's 6 copies done
    } else {
      asm volatile("s_wait_asynccnt 0x0" ::: "memory");
    }
    __syncthreads();   // LDS[buf] visible to all waves

    union Fr { uint4 q[2]; v16bf v; };
    v16bf afr[4], bfr[4];
#pragma unroll
    for (int i = 0; i < 4; ++i) {
      const unsigned short* base = &Xs[buf][(waveM * 64 + i * 16 + frow) * LSTR];
      Fr f;
      f.q[0] = *reinterpret_cast<const uint4*>(base + akb);
      f.q[1] = *reinterpret_cast<const uint4*>(base + akb + 16);
      afr[i] = f.v;
    }
#pragma unroll
    for (int j = 0; j < 4; ++j) {
      const unsigned short* base = &Ws[buf][(waveN * 64 + j * 16 + frow) * LSTR];
      Fr f;
      f.q[0] = *reinterpret_cast<const uint4*>(base + bkb);
      f.q[1] = *reinterpret_cast<const uint4*>(base + bkb + 16);
      bfr[j] = f.v;
    }

#pragma unroll
    for (int i = 0; i < 4; ++i)
#pragma unroll
      for (int j = 0; j < 4; ++j)
        acc[i][j] = __builtin_amdgcn_wmma_f32_16x16x32_bf16(
            false, afr[i], false, bfr[j], (short)0, acc[i][j], false, false);

    __syncthreads();   // all waves done reading LDS[buf] before async overwrites it
    buf ^= 1;
  }

  // epilogue: v8f element e -> M = e + 8*(lane>>4), N = lane&15
  const int mhi      = (lane >> 4) * 8;
  const int ncolBase = nCTA + waveN * 64 + (lane & 15);
#pragma unroll
  for (int j = 0; j < 4; ++j) {
    const int ncol = ncolBase + j * 16;
    const float bb = bias[ncol];
#pragma unroll
    for (int i = 0; i < 4; ++i) {
      const int mbase = mCTA + waveM * 64 + i * 16 + mhi;
#pragma unroll
      for (int e = 0; e < 8; ++e)
        out[(size_t)(mbase + e) * OUT_F + ncol] = acc[i][j][e] + bb;
    }
  }
}

extern "C" void kernel_launch(void* const* d_in, const int* in_sizes, int n_in,
                              void* d_out, int out_size, void* d_ws, size_t ws_size,
                              hipStream_t stream) {
  (void)in_sizes; (void)n_in; (void)out_size; (void)ws_size;
  const float* x    = (const float*)d_in[0];
  const float* W    = (const float*)d_in[1];
  const float* A    = (const float*)d_in[2];
  const float* B    = (const float*)d_in[3];
  const float* bias = (const float*)d_in[4];
  float* out = (float*)d_out;

  unsigned short* xb = (unsigned short*)d_ws;                 // 8192*4096 bf16 (64 MiB)
  unsigned short* wb = xb + (size_t)MTOT * IN_F;              // 4096*4096 bf16 (32 MiB)

  const int n8 = MTOT * IN_F / 8;
  cvt_x<<<(n8 + 255) / 256, 256, 0, stream>>>(x, xb, n8);
  fuse_w<<<(OUT_F * IN_F) / 256, 256, 0, stream>>>(W, A, B, wb);

  dim3 grid(OUT_F / 256, MTOT / 128);
  lora_gemm<<<grid, dim3(256), 0, stream>>>(xb, wb, bias, out);
}
